// SupConLoss_33844342293046
// MI455X (gfx1250) — matrix operand order repaired and checked by reading, not compiled, not run on hardware
//
#include <hip/hip_runtime.h>
#include <hip/hip_fp16.h>
#include <math.h>

typedef __attribute__((ext_vector_type(16))) _Float16 v16h;
typedef __attribute__((ext_vector_type(8)))  float    v8f;

union Frag16 {
    v16h   h;
    float4 q[2];   // two 16-byte halves of the 32-byte fragment
};

// 2*log2(e): arg for v_exp_f32 (2^x) so that 2^(K*dot) == exp(+-dot/T), T=0.5
#define TWO_LOG2E 2.8853900817779268f

// ---------------------------------------------------------------------------
// Kernel 1: L2-normalize rows (fp32 in), emit f16 normalized rows + is_normal
// One wave (32 lanes) per row; D == 128 -> 4 floats per lane.
// ---------------------------------------------------------------------------
__global__ __launch_bounds__(256)
void supcon_normalize(const float* __restrict__ feat,
                      const int*   __restrict__ labels,
                      const int*   __restrict__ normal_label,
                      _Float16*    __restrict__ nf,
                      float*       __restrict__ isnf,
                      int N, int D)
{
    const int row  = (blockIdx.x * blockDim.x + threadIdx.x) >> 5;
    const int lane = threadIdx.x & 31;
    if (row >= N) return;

    const float* rp = feat + (size_t)row * D;
    float4 v = *reinterpret_cast<const float4*>(rp + lane * 4);

    float ss = v.x * v.x + v.y * v.y + v.z * v.z + v.w * v.w;
    #pragma unroll
    for (int m = 1; m < 32; m <<= 1) ss += __shfl_xor(ss, m, 32);

    const float inv = 1.0f / fmaxf(sqrtf(ss), 1e-12f);

    union { _Float16 h[4]; float2 f2; } o;
    o.h[0] = (_Float16)(v.x * inv);
    o.h[1] = (_Float16)(v.y * inv);
    o.h[2] = (_Float16)(v.z * inv);
    o.h[3] = (_Float16)(v.w * inv);
    *reinterpret_cast<float2*>(nf + (size_t)row * D + lane * 4) = o.f2;

    if (lane == 0)
        isnf[row] = (labels[row] == normal_label[0]) ? 1.0f : 0.0f;
}

// ---------------------------------------------------------------------------
// Kernel 2: fused WMMA GEMM (NF * NF^T / T) + masked exp row-reductions.
// Block = 8 waves owning 32 rows (two 16-row strips sharing B fragments);
// waves stripe the j-tiles. 8 v_wmma per 8 B-loads (K = 128, 4 steps).
// ---------------------------------------------------------------------------
__global__ __launch_bounds__(256)
void supcon_main(const _Float16* __restrict__ nf,
                 const float*    __restrict__ isnf,
                 float*          __restrict__ term,
                 int N, int D)
{
    const int lane = threadIdx.x & 31;
    const int wave = threadIdx.x >> 5;   // 0..7
    const int half = lane >> 4;          // 0 or 1
    const int l16  = lane & 15;
    const int i0   = blockIdx.x * 32;    // 32 rows per block
    const int NT   = N >> 4;             // number of 16-wide j tiles

    // A fragments for two row strips: lane holds rows i0+l16 and i0+16+l16.
    // Per ISA: elems 0..7 = K[half*8,+8), elems 8..15 = K[16+half*8,+8).
    const _Float16* ar0 = nf + (size_t)(i0 + l16) * D;
    const _Float16* ar1 = nf + (size_t)(i0 + 16 + l16) * D;
    Frag16 a0[4], a1[4];
    #pragma unroll
    for (int kk = 0; kk < 4; ++kk) {
        const int kb = kk * 32;
        a0[kk].q[0] = *reinterpret_cast<const float4*>(ar0 + kb + half * 8);
        a0[kk].q[1] = *reinterpret_cast<const float4*>(ar0 + kb + 16 + half * 8);
        a1[kk].q[0] = *reinterpret_cast<const float4*>(ar1 + kb + half * 8);
        a1[kk].q[1] = *reinterpret_cast<const float4*>(ar1 + kb + 16 + half * 8);
    }

    float pos0[8], neg0[8], pos1[8], neg1[8];
    #pragma unroll
    for (int r = 0; r < 8; ++r) {
        pos0[r] = 0.0f; neg0[r] = 0.0f;
        pos1[r] = 0.0f; neg1[r] = 0.0f;
    }

    for (int jt = wave; jt < NT; jt += 8) {
        const int j0 = jt * 16;
        // B fragment: lane holds column j0 + l16, elems 0..7 = K[half*8,+8), elems 8..15 = K[16+half*8,+8).
        const _Float16* brow = nf + (size_t)(j0 + l16) * D;

        v8f c0 = {};
        v8f c1 = {};
        #pragma unroll
        for (int kk = 0; kk < 4; ++kk) {
            Frag16 b;
            const int kb = kk * 32 + half * 8;
            b.q[0] = *reinterpret_cast<const float4*>(brow + kb);
            b.q[1] = *reinterpret_cast<const float4*>(brow + kb + 16);
            c0 = __builtin_amdgcn_wmma_f32_16x16x32_f16(
                     false, a0[kk].h, false, b.h, (short)0, c0, false, false);
            c1 = __builtin_amdgcn_wmma_f32_16x16x32_f16(
                     false, a1[kk].h, false, b.h, (short)0, c1, false, false);
        }

        // Column mask is per-lane constant for this tile.
        const float jn = isnf[j0 + l16];                  // 1 if column normal
        // K = (2*jn - 1) * 2*log2(e): 2^(K*dot) = exp(+-dot / T)
        const float K  = fmaf(jn, 2.0f * TWO_LOG2E, -TWO_LOG2E);
        const float nj = 1.0f - jn;
        #pragma unroll
        for (int r = 0; r < 8; ++r) {
            const float e0 = __builtin_amdgcn_exp2f(c0[r] * K);
            const float e1 = __builtin_amdgcn_exp2f(c1[r] * K);
            pos0[r] += jn * e0;  neg0[r] += nj * e0;
            pos1[r] += jn * e1;  neg1[r] += nj * e1;
        }
    }

    // Reduce each row accumulator across the 16 lanes of each half-wave.
    #pragma unroll
    for (int r = 0; r < 8; ++r) {
        float p0 = pos0[r], n0 = neg0[r], p1 = pos1[r], n1 = neg1[r];
        #pragma unroll
        for (int m = 1; m < 16; m <<= 1) {
            p0 += __shfl_xor(p0, m, 32);
            n0 += __shfl_xor(n0, m, 32);
            p1 += __shfl_xor(p1, m, 32);
            n1 += __shfl_xor(n1, m, 32);
        }
        pos0[r] = p0; neg0[r] = n0; pos1[r] = p1; neg1[r] = n1;
    }

    __shared__ float ldsP[8][32];
    __shared__ float ldsN[8][32];
    if (l16 == 0) {
        #pragma unroll
        for (int r = 0; r < 8; ++r) {
            ldsP[wave][r + 8 * half]      = pos0[r];
            ldsN[wave][r + 8 * half]      = neg0[r];
            ldsP[wave][16 + r + 8 * half] = pos1[r];
            ldsN[wave][16 + r + 8 * half] = neg1[r];
        }
    }
    __syncthreads();

    if (threadIdx.x < 32) {
        const int row = threadIdx.x;
        float p = 0.0f, n = 0.0f;
        #pragma unroll
        for (int w = 0; w < 8; ++w) { p += ldsP[w][row]; n += ldsN[w][row]; }
        const float in = isnf[i0 + row];   // row mask factors out of both sums
        p *= in; n *= in;
        term[i0 + row] = -(logf(p + 1e-8f) + logf(n + 1e-8f));
    }
}

// ---------------------------------------------------------------------------
// Kernel 3: reduce per-row terms to the scalar loss (deterministic, 1 block)
// ---------------------------------------------------------------------------
__global__ __launch_bounds__(256)
void supcon_reduce(const float* __restrict__ term, float* __restrict__ out, int N)
{
    __shared__ float s[256];
    float acc = 0.0f;
    for (int i = threadIdx.x; i < N; i += 256) acc += term[i];
    s[threadIdx.x] = acc;
    __syncthreads();
    for (int off = 128; off > 0; off >>= 1) {
        if (threadIdx.x < off) s[threadIdx.x] += s[threadIdx.x + off];
        __syncthreads();
    }
    if (threadIdx.x == 0) out[0] = s[0] / (float)N;
}

extern "C" void kernel_launch(void* const* d_in, const int* in_sizes, int n_in,
                              void* d_out, int out_size, void* d_ws, size_t ws_size,
                              hipStream_t stream)
{
    const float* feat   = (const float*)d_in[0];
    const int*   labels = (const int*)d_in[1];
    const int*   normal = (const int*)d_in[2];

    const int N = in_sizes[1];            // 8192
    const int D = in_sizes[0] / N;        // 128

    char* ws = (char*)d_ws;
    _Float16* nf      = (_Float16*)ws;                         // N*D f16 (2 MB)
    size_t    nfBytes = (size_t)N * D * sizeof(_Float16);
    float*    isnf    = (float*)(ws + nfBytes);                // N floats
    float*    term    = (float*)(ws + nfBytes + (size_t)N * sizeof(float));

    // 1 wave per row -> N waves -> N*32 threads
    supcon_normalize<<<(N * 32 + 255) / 256, 256, 0, stream>>>(
        feat, labels, normal, nf, isnf, N, D);

    // 32 rows per block
    supcon_main<<<N / 32, 256, 0, stream>>>(nf, isnf, term, N, D);

    supcon_reduce<<<1, 256, 0, stream>>>(term, (float*)d_out, N);
}
